// TransformerOCModel_28544352649251
// MI455X (gfx1250) — matrix operand order repaired and checked by reading, not compiled, not run on hardware
//
#include <hip/hip_runtime.h>
#include <hip/hip_bf16.h>

typedef __attribute__((ext_vector_type(16))) _Float16 v16h;
typedef __attribute__((ext_vector_type(8)))  _Float16 v8h;
typedef __attribute__((ext_vector_type(8)))  float    v8f;

#define NPTS 8192
#define HID  256
#define KNE  64
#define DRTHR 0.2f
#define CAP  2048
#define KT   64

// denorm constants for columns {13,14,15, 22,23,24, 40,41,42}
__constant__ float c_mn[9] = {-73.8012f,-72.4212f,-187.575f,
                              -93.8255f,-93.9099f,-223.854f,
                              -110.016f,-110.015f,-267.235f};
__constant__ float c_mx[9] = { 73.7524f, 70.415f, 187.575f,
                               93.8869f, 93.8406f, 223.854f,
                               110.0128f,110.015f, 267.235f};

// ---------------------------------------------------------------------------
// CDNA5 async-copy helpers (GLOBAL_LOAD_ASYNC_TO_LDS_B128, ASYNCcnt fence).
// LDS operand = LDS byte offset = low 32 bits of the generic pointer
// (flat LDS aperture keeps the offset in addr[31:0]).
// ---------------------------------------------------------------------------
__device__ __forceinline__ unsigned lds_off(const void* p) {
  return (unsigned)(unsigned long long)p;
}
__device__ __forceinline__ void async_b128(unsigned lds, const void* gaddr) {
  asm volatile("global_load_async_to_lds_b128 %0, %1, off"
               :: "v"(lds), "v"(gaddr) : "memory");
}
__device__ __forceinline__ void wait_async0() {
  asm volatile("s_wait_asynccnt 0" ::: "memory");
}

// ---------------------------------------------------------------------------
// Kernel 1: per-node (r, eta, sin(phi), cos(phi)) for 3 prongs
// ---------------------------------------------------------------------------
__global__ void coords_kernel(const float* __restrict__ xr,
                              float* __restrict__ coords) {
  int n = blockIdx.x * blockDim.x + threadIdx.x;
  if (n >= NPTS) return;
  const int colx[3] = {13, 22, 40};
  #pragma unroll
  for (int p = 0; p < 3; ++p) {
    int c = colx[p];
    float ax = xr[n*49 + c + 0] * (c_mx[p*3+0] - c_mn[p*3+0] + 1e-8f) + c_mn[p*3+0];
    float ay = xr[n*49 + c + 1] * (c_mx[p*3+1] - c_mn[p*3+1] + 1e-8f) + c_mn[p*3+1];
    float az = xr[n*49 + c + 2] * (c_mx[p*3+2] - c_mn[p*3+2] + 1e-8f) + c_mn[p*3+2];
    float r     = sqrtf(ax*ax + ay*ay);
    float phi   = atan2f(ay, ax);
    float theta = atan2f(r, az);
    float t     = fmaxf(tanf(0.5f * theta), 1e-6f);
    float eta   = -logf(t);
    coords[n*12 + 0 + p] = r;
    coords[n*12 + 3 + p] = eta;
    coords[n*12 + 6 + p] = sinf(phi);
    coords[n*12 + 9 + p] = cosf(phi);
  }
}

// ---------------------------------------------------------------------------
// Kernel 2: neighbor build (LDS atomic candidate buffer + 64-round argmin,
// tie-break on smaller j for atomic-order independence)
// ---------------------------------------------------------------------------
__global__ __launch_bounds__(256) void neigh_kernel(
    const float* __restrict__ coords, const int* __restrict__ batch,
    int* __restrict__ nidx, int* __restrict__ nmask) {
  __shared__ float s_ri[3], s_ei[3], s_si[3], s_ci[3];
  __shared__ int   s_cnt;
  __shared__ float s_d[CAP];
  __shared__ int   s_j[CAP];
  __shared__ float r_v[256];
  __shared__ int   r_j[256];
  __shared__ int   r_p[256];
  const int i = blockIdx.x, tid = threadIdx.x;
  if (tid < 3) {
    s_ri[tid] = coords[i*12 + 0 + tid];
    s_ei[tid] = coords[i*12 + 3 + tid];
    s_si[tid] = coords[i*12 + 6 + tid];
    s_ci[tid] = coords[i*12 + 9 + tid];
  }
  if (tid == 0) s_cnt = 0;
  __syncthreads();
  const int bi = batch[i];
  for (int j = tid; j < NPTS; j += 256) {
    float rj0 = coords[j*12+0], rj1 = coords[j*12+1], rj2 = coords[j*12+2];
    float best = 1e30f; int ba = 0, bb = 0;
    #pragma unroll
    for (int a = 0; a < 3; ++a) {
      float ra = s_ri[a];
      float d0 = fabsf(ra - rj0), d1 = fabsf(ra - rj1), d2 = fabsf(ra - rj2);
      if (d0 < best) { best = d0; ba = a; bb = 0; }
      if (d1 < best) { best = d1; ba = a; bb = 1; }
      if (d2 < best) { best = d2; ba = a; bb = 2; }
    }
    float ej = coords[j*12+3+bb], sj = coords[j*12+6+bb], cj = coords[j*12+9+bb];
    float deta = s_ei[ba] - ej;
    float sd = s_si[ba]*cj - s_ci[ba]*sj;
    float cd = s_ci[ba]*cj + s_si[ba]*sj;
    float dphi = atan2f(sd, cd);
    float dr = sqrtf(deta*deta + dphi*dphi);
    if (batch[j] != bi) dr += 1.0e6f;
    if (dr <= DRTHR) {
      int p = atomicAdd(&s_cnt, 1);
      if (p < CAP) { s_d[p] = dr; s_j[p] = j; }
    }
  }
  __syncthreads();
  int cnt = s_cnt < CAP ? s_cnt : CAP;
  for (int kk = 0; kk < KNE; ++kk) {
    float bv = __builtin_inff(); int bj = 0x7fffffff; int bp = -1;
    for (int p = tid; p < cnt; p += 256) {
      float v = s_d[p]; int jv = s_j[p];
      if (v <= DRTHR && (v < bv || (v == bv && jv < bj))) { bv = v; bj = jv; bp = p; }
    }
    r_v[tid] = bv; r_j[tid] = bj; r_p[tid] = bp;
    __syncthreads();
    for (int s = 128; s > 0; s >>= 1) {
      if (tid < s) {
        float v2 = r_v[tid+s]; int j2 = r_j[tid+s];
        if (v2 < r_v[tid] || (v2 == r_v[tid] && j2 < r_j[tid])) {
          r_v[tid] = v2; r_j[tid] = j2; r_p[tid] = r_p[tid+s];
        }
      }
      __syncthreads();
    }
    if (tid == 0) {
      if (r_p[0] >= 0) {
        nidx[i*KNE + kk]  = r_j[0];
        nmask[i*KNE + kk] = 0;
        s_d[r_p[0]] = __builtin_inff();
      } else {
        nidx[i*KNE + kk]  = i;
        nmask[i*KNE + kk] = (kk == 0 && cnt == 0) ? 0 : 1;
      }
    }
    __syncthreads();
  }
}

// ---------------------------------------------------------------------------
// Kernel 3: input projection x = x_raw @ W_in + b_in  (K=49 -> plain VALU)
// ---------------------------------------------------------------------------
__global__ void inproj_kernel(const float* __restrict__ xr,
                              const float* __restrict__ W,
                              const float* __restrict__ b,
                              float* __restrict__ x) {
  int idx = blockIdx.x * 256 + threadIdx.x;
  int m = idx >> 8, c = idx & 255;
  float acc = b[c];
  #pragma unroll 7
  for (int t = 0; t < 49; ++t) acc += xr[m*49 + t] * W[t*256 + c];
  x[idx] = acc;
}

// ---------------------------------------------------------------------------
// Kernel 4: weight convert+transpose  Wt[n][k] = (f16) W[k][n]
// ---------------------------------------------------------------------------
__global__ void convert_wt_kernel(const float* __restrict__ W,
                                  _Float16* __restrict__ Wt, int K, int N) {
  int idx = blockIdx.x * 256 + threadIdx.x;
  if (idx >= K * N) return;
  int n = idx / K, k = idx % K;
  Wt[idx] = (_Float16)W[(size_t)k * N + n];
}

// ---------------------------------------------------------------------------
// Kernel 5: WMMA GEMM, f16 inputs.  C = op(A[M,K] @ Bt[N,K]^T + bias (+res)).
// 128x64 block tile, 8 waves x (2x2) 16x16x32 WMMA subtiles, KT=64 stages,
// double-buffered LDS filled with GLOBAL_LOAD_ASYNC_TO_LDS_B128.
// flags bit0 = ReLU.  C16 (optional) = f16 copy of the output.
// ---------------------------------------------------------------------------
__global__ __launch_bounds__(256) void gemm_wmma(
    const _Float16* __restrict__ A,  const _Float16* __restrict__ Bt,
    const float* __restrict__ bias,  const float* __restrict__ res,
    float* __restrict__ C, _Float16* __restrict__ C16,
    int M, int N, int K, int flags) {
  __shared__ alignas(16) _Float16 As[2][128 * KT];
  __shared__ alignas(16) _Float16 Bs[2][64 * KT];
  const int tid  = threadIdx.x;
  const int lane = tid & 31, wid = tid >> 5;
  const int wm = wid & 3, wn = wid >> 2;
  const int lrow = lane & 15, lhi = lane >> 4;
  const int bm0 = blockIdx.y * 128;
  const int bn0 = blockIdx.x * 64;

  v8f acc[2][2];
  #pragma unroll
  for (int i = 0; i < 2; ++i)
    #pragma unroll
    for (int j = 0; j < 2; ++j)
      #pragma unroll
      for (int e = 0; e < 8; ++e) acc[i][j][e] = 0.0f;

  // A tile: 128 rows x 128B = 1024 16B-chunks (4/thread)
  // B tile:  64 rows x 128B =  512 16B-chunks (2/thread)
  auto issue_tile = [&](int buf, int kb) {
    #pragma unroll
    for (int c = 0; c < 4; ++c) {
      int ch = tid * 4 + c;
      int r = ch >> 3, col = (ch & 7) * 8;
      async_b128(lds_off(&As[buf][r * KT + col]),
                 A + (size_t)(bm0 + r) * K + kb + col);
    }
    #pragma unroll
    for (int c = 0; c < 2; ++c) {
      int ch = tid * 2 + c;
      int r = ch >> 3, col = (ch & 7) * 8;
      async_b128(lds_off(&Bs[buf][r * KT + col]),
                 Bt + (size_t)(bn0 + r) * K + kb + col);
    }
  };

  issue_tile(0, 0);
  wait_async0();
  __syncthreads();

  const int nk = K / KT;
  for (int t = 0; t < nk; ++t) {
    const int cur = t & 1;
    if (t + 1 < nk) issue_tile(1 - cur, (t + 1) * KT);
    const _Float16* Ab = &As[cur][0];
    const _Float16* Bb = &Bs[cur][0];
    #pragma unroll
    for (int ks = 0; ks < 2; ++ks) {
      v16h af[2], bf[2];
      #pragma unroll
      for (int sm = 0; sm < 2; ++sm) {
        const _Float16* ap = &Ab[(wm*32 + sm*16 + lrow) * KT + ks*32 + lhi*16];
        v8h lo = *(const v8h*)ap;
        v8h hi = *(const v8h*)(ap + 8);
        af[sm] = __builtin_shufflevector(lo, hi,
                   0,1,2,3,4,5,6,7,8,9,10,11,12,13,14,15);
      }
      #pragma unroll
      for (int sn = 0; sn < 2; ++sn)
        bf[sn] = *(const v16h*)&Bb[(wn*32 + sn*16 + lrow) * KT + ks*32 + lhi*16];
      #pragma unroll
      for (int sm = 0; sm < 2; ++sm)
        #pragma unroll
        for (int sn = 0; sn < 2; ++sn)
          acc[sm][sn] = __builtin_amdgcn_wmma_f32_16x16x32_f16(
              false, af[sm], false, bf[sn], (short)0, acc[sm][sn], false, false);
    }
    wait_async0();
    __syncthreads();
  }

  const bool relu = flags & 1;
  #pragma unroll
  for (int sm = 0; sm < 2; ++sm) {
    #pragma unroll
    for (int sn = 0; sn < 2; ++sn) {
      int gm0 = bm0 + wm*32 + sm*16 + lhi*8;   // C layout: M = r + 8*lane[4]
      int gn  = bn0 + wn*32 + sn*16 + lrow;    //           N = lane[3:0]
      float bvv = bias ? bias[gn] : 0.0f;
      #pragma unroll
      for (int r = 0; r < 8; ++r) {
        int gm = gm0 + r;
        float v = acc[sm][sn][r] + bvv;
        if (relu) v = fmaxf(v, 0.0f);
        if (res)  v += res[(size_t)gm * N + gn];
        C[(size_t)gm * N + gn] = v;
        if (C16) C16[(size_t)gm * N + gn] = (_Float16)v;
      }
    }
  }
}

// ---------------------------------------------------------------------------
// Kernel 6: layernorm -> f16 output (feeds WMMA GEMMs)
// ---------------------------------------------------------------------------
__global__ __launch_bounds__(256) void ln_kernel(
    const float* __restrict__ x, const float* __restrict__ g,
    const float* __restrict__ b, _Float16* __restrict__ y16) {
  __shared__ float red[256];
  int row = blockIdx.x, tid = threadIdx.x;
  float v = x[(size_t)row*HID + tid];
  red[tid] = v; __syncthreads();
  for (int s = 128; s > 0; s >>= 1) { if (tid < s) red[tid] += red[tid+s]; __syncthreads(); }
  float mu = red[0] * (1.0f/256.0f);
  __syncthreads();
  float d = v - mu;
  red[tid] = d*d; __syncthreads();
  for (int s = 128; s > 0; s >>= 1) { if (tid < s) red[tid] += red[tid+s]; __syncthreads(); }
  float var = red[0] * (1.0f/256.0f);
  y16[(size_t)row*HID + tid] = (_Float16)(d * rsqrtf(var + 1e-5f) * g[tid] + b[tid]);
}

// ---------------------------------------------------------------------------
// Kernel 7: sparse attention (f32 q/k/v, f16 output feeding Wo GEMM)
// ---------------------------------------------------------------------------
__global__ __launch_bounds__(256) void attn_kernel(
    const float* __restrict__ q, const float* __restrict__ kall,
    const float* __restrict__ vall, const int* __restrict__ nidx,
    const int* __restrict__ nmask, _Float16* __restrict__ out16) {
  __shared__ float s_q[HID];
  __shared__ float s_sc[8*64];
  __shared__ int   s_i[64];
  __shared__ int   s_m[64];
  const int n = blockIdx.x, tid = threadIdx.x;
  s_q[tid] = q[(size_t)n*HID + tid];
  if (tid < 64) { s_i[tid] = nidx[n*64 + tid]; s_m[tid] = nmask[n*64 + tid]; }
  __syncthreads();
  const int h = tid >> 5, sub = tid & 31;
  const float scale = 0.17677669529663689f;   // 32^-0.5
  #pragma unroll
  for (int rep = 0; rep < 2; ++rep) {
    int kk = sub + rep*32;
    float sc;
    if (s_m[kk]) sc = -__builtin_inff();
    else {
      const float* kr = kall + (size_t)s_i[kk]*HID + h*32;
      float a = 0.f;
      #pragma unroll
      for (int d = 0; d < 32; ++d) a += s_q[h*32 + d] * kr[d];
      sc = a * scale;
    }
    s_sc[h*64 + kk] = sc;
  }
  __syncthreads();
  if (tid < 8) {
    float m = -__builtin_inff();
    for (int kk = 0; kk < 64; ++kk) m = fmaxf(m, s_sc[tid*64 + kk]);
    float ssum = 0.f;
    for (int kk = 0; kk < 64; ++kk) {
      float sc = s_sc[tid*64 + kk];
      float e = (sc == -__builtin_inff()) ? 0.f : __expf(sc - m);
      s_sc[tid*64 + kk] = e;
      ssum += e;
    }
    float inv = (ssum > 0.f) ? 1.f/ssum : 0.f;   // all-masked -> zeros
    for (int kk = 0; kk < 64; ++kk) s_sc[tid*64 + kk] *= inv;
  }
  __syncthreads();
  float a = 0.f;
  for (int kk = 0; kk < 64; ++kk) {
    float w = s_sc[h*64 + kk];
    if (w != 0.f) a += w * vall[(size_t)s_i[kk]*HID + h*32 + sub];
  }
  out16[(size_t)n*HID + tid] = (_Float16)a;
}

// ---------------------------------------------------------------------------
// Head kernels
// ---------------------------------------------------------------------------
__global__ void lat2_kernel(const float* __restrict__ h, const float* __restrict__ W,
                            const float* __restrict__ b, float* __restrict__ lat) {
  int idx = blockIdx.x*256 + threadIdx.x;
  if (idx >= NPTS*24) return;
  int m = idx / 24, c = idx % 24;
  float acc = b[c];
  for (int t = 0; t < HID; ++t) acc += h[(size_t)m*HID + t] * W[t*24 + c];
  lat[idx] = acc;
}

__global__ void latnorm_kernel(const float* __restrict__ lat, float* __restrict__ out) {
  int n = blockIdx.x*256 + threadIdx.x;
  if (n >= NPTS) return;
  float ss = 0.f;
  #pragma unroll
  for (int c = 0; c < 24; ++c) { float v = lat[n*24 + c]; ss += v*v; }
  float inv = 1.f / fmaxf(sqrtf(ss), 1e-12f);
  #pragma unroll
  for (int c = 0; c < 24; ++c) out[n*24 + c] = lat[n*24 + c] * inv;
}

__global__ void beta_kernel(const float* __restrict__ h, const float* __restrict__ W,
                            const float* __restrict__ b, float* __restrict__ out) {
  int n = blockIdx.x*256 + threadIdx.x;
  if (n >= NPTS) return;
  float acc = b[0];
  for (int t = 0; t < 128; ++t) acc += h[(size_t)n*128 + t] * W[t];
  float bv = 1.f / (1.f + __expf(-acc));
  out[n] = fminf(fmaxf(bv, 1e-6f), 1.f - 1e-6f);
}

// ---------------------------------------------------------------------------
// Host orchestration
// ---------------------------------------------------------------------------
extern "C" void kernel_launch(void* const* d_in, const int* in_sizes, int n_in,
                              void* d_out, int out_size, void* d_ws, size_t ws_size,
                              hipStream_t stream) {
  (void)in_sizes; (void)n_in; (void)out_size; (void)ws_size;
  const float* x_raw = (const float*)d_in[0];
  const int*   batch = (const int*)  d_in[1];
  const float* W_in  = (const float*)d_in[2];
  const float* b_in  = (const float*)d_in[3];
  const float* ln1g  = (const float*)d_in[4];
  const float* ln1b  = (const float*)d_in[5];
  const float* ln2g  = (const float*)d_in[6];
  const float* ln2b  = (const float*)d_in[7];
  const float* Wq    = (const float*)d_in[8];
  const float* Wk    = (const float*)d_in[9];
  const float* Wv    = (const float*)d_in[10];
  const float* Wo    = (const float*)d_in[11];
  const float* bo    = (const float*)d_in[12];
  const float* Wf1   = (const float*)d_in[13];
  const float* bf1   = (const float*)d_in[14];
  const float* Wf2   = (const float*)d_in[15];
  const float* bf2   = (const float*)d_in[16];
  const float* Wl1   = (const float*)d_in[17];
  const float* bl1   = (const float*)d_in[18];
  const float* Wl2   = (const float*)d_in[19];
  const float* bl2   = (const float*)d_in[20];
  const float* Wb1   = (const float*)d_in[21];
  const float* bb1   = (const float*)d_in[22];
  const float* Wb2   = (const float*)d_in[23];
  const float* bb2   = (const float*)d_in[24];

  // ---- workspace layout (f32/int region, then f16 region) ----
  float* ws     = (float*)d_ws;
  float* coords = ws;                              // 8192*12
  int*   nidx   = (int*)(coords + NPTS*12);        // 8192*64
  int*   nmask  = nidx + NPTS*KNE;                 // 8192*64
  float* x      = (float*)(nmask + NPTS*KNE);      // 8192*256
  float* lnb    = x   + (size_t)NPTS*HID;          // head scratch (f32)
  float* qb     = lnb + (size_t)NPTS*HID;
  float* kb     = qb  + (size_t)NPTS*HID;
  float* vb     = kb  + (size_t)NPTS*HID;
  float* fh     = vb  + (size_t)NPTS*HID;          // 8192*512

  _Float16* h16   = (_Float16*)(fh + (size_t)NPTS*512);
  _Float16* x16   = h16;                           // 8192*256
  _Float16* lnb16 = x16   + (size_t)NPTS*HID;      // 8192*256
  _Float16* ab16  = lnb16 + (size_t)NPTS*HID;      // 8192*256
  _Float16* fh16  = ab16  + (size_t)NPTS*HID;      // 8192*512
  _Float16* wtq   = fh16  + (size_t)NPTS*512;      // 4*256*256
  _Float16* wtk   = wtq  + (size_t)4*HID*HID;
  _Float16* wtv   = wtk  + (size_t)4*HID*HID;
  _Float16* wto   = wtv  + (size_t)4*HID*HID;
  _Float16* wtf1  = wto  + (size_t)4*HID*HID;      // 4*512*256 ([N=512][K=256])
  _Float16* wtf2  = wtf1 + (size_t)4*HID*512;      // 4*256*512
  _Float16* wtl1  = wtf2 + (size_t)4*HID*512;      // 256*256
  _Float16* wtb1  = wtl1 + (size_t)HID*HID;        // 128*256

  float* beta_out = (float*)d_out;
  float* lat_out  = beta_out + NPTS;

  // ---- weight convert+transpose (f32 [K][N] -> f16 [N][K]) ----
  for (int l = 0; l < 4; ++l) {
    convert_wt_kernel<<<(HID*HID+255)/256, 256, 0, stream>>>(Wq + (size_t)l*HID*HID, wtq + (size_t)l*HID*HID, HID, HID);
    convert_wt_kernel<<<(HID*HID+255)/256, 256, 0, stream>>>(Wk + (size_t)l*HID*HID, wtk + (size_t)l*HID*HID, HID, HID);
    convert_wt_kernel<<<(HID*HID+255)/256, 256, 0, stream>>>(Wv + (size_t)l*HID*HID, wtv + (size_t)l*HID*HID, HID, HID);
    convert_wt_kernel<<<(HID*HID+255)/256, 256, 0, stream>>>(Wo + (size_t)l*HID*HID, wto + (size_t)l*HID*HID, HID, HID);
    convert_wt_kernel<<<(HID*512+255)/256, 256, 0, stream>>>(Wf1 + (size_t)l*HID*512, wtf1 + (size_t)l*HID*512, HID, 512);
    convert_wt_kernel<<<(HID*512+255)/256, 256, 0, stream>>>(Wf2 + (size_t)l*HID*512, wtf2 + (size_t)l*HID*512, 512, HID);
  }
  convert_wt_kernel<<<(HID*HID+255)/256, 256, 0, stream>>>(Wl1, wtl1, HID, HID);
  convert_wt_kernel<<<(HID*128+255)/256, 256, 0, stream>>>(Wb1, wtb1, HID, 128);

  coords_kernel<<<NPTS/256, 256, 0, stream>>>(x_raw, coords);
  neigh_kernel<<<NPTS, 256, 0, stream>>>(coords, batch, nidx, nmask);
  inproj_kernel<<<NPTS*HID/256, 256, 0, stream>>>(x_raw, W_in, b_in, x);

  dim3 blk(256);
  dim3 g256(HID/64, NPTS/128);
  dim3 g512(512/64, NPTS/128);
  for (int l = 0; l < 4; ++l) {
    const _Float16* wq = wtq + (size_t)l*HID*HID;
    const _Float16* wk = wtk + (size_t)l*HID*HID;
    const _Float16* wv = wtv + (size_t)l*HID*HID;
    const _Float16* wo = wto + (size_t)l*HID*HID;
    const _Float16* wf1 = wtf1 + (size_t)l*HID*512;
    const _Float16* wf2 = wtf2 + (size_t)l*HID*512;

    ln_kernel<<<NPTS, blk, 0, stream>>>(x, ln1g + l*HID, ln1b + l*HID, lnb16);
    gemm_wmma<<<g256, blk, 0, stream>>>(lnb16, wq, nullptr, nullptr, qb, nullptr, NPTS, HID, HID, 0);
    gemm_wmma<<<g256, blk, 0, stream>>>(lnb16, wk, nullptr, nullptr, kb, nullptr, NPTS, HID, HID, 0);
    gemm_wmma<<<g256, blk, 0, stream>>>(lnb16, wv, nullptr, nullptr, vb, nullptr, NPTS, HID, HID, 0);
    attn_kernel<<<NPTS, blk, 0, stream>>>(qb, kb, vb, nidx, nmask, ab16);
    gemm_wmma<<<g256, blk, 0, stream>>>(ab16, wo, bo + l*HID, x, x, nullptr, NPTS, HID, HID, 0);
    ln_kernel<<<NPTS, blk, 0, stream>>>(x, ln2g + l*HID, ln2b + l*HID, lnb16);
    gemm_wmma<<<g512, blk, 0, stream>>>(lnb16, wf1, bf1 + l*512, nullptr, fh, fh16, NPTS, 512, HID, 1);
    gemm_wmma<<<g256, blk, 0, stream>>>(fh16, wf2, bf2 + l*HID, x, x,
                                        (l == 3) ? x16 : nullptr, NPTS, HID, 512, 0);
  }

  // heads
  gemm_wmma<<<g256, blk, 0, stream>>>(x16, wtl1, bl1, nullptr, lnb, nullptr, NPTS, HID, HID, 1);
  lat2_kernel<<<(NPTS*24 + 255)/256, 256, 0, stream>>>(lnb, Wl2, bl2, qb);
  latnorm_kernel<<<NPTS/256, 256, 0, stream>>>(qb, lat_out);

  gemm_wmma<<<dim3(128/64, NPTS/128), blk, 0, stream>>>(x16, wtb1, bb1, nullptr, kb, nullptr, NPTS, 128, HID, 1);
  beta_kernel<<<NPTS/256, 256, 0, stream>>>(kb, Wb2, bb2, beta_out);
}
